// OneRecTokenizer_3229815407338
// MI455X (gfx1250) — matrix operand order repaired and checked by reading, not compiled, not run on hardware
//
#include <hip/hip_runtime.h>

// ---------------- types ----------------
typedef __attribute__((ext_vector_type(16))) __bf16        v16bf;
typedef __attribute__((ext_vector_type(8)))  float         v8f;
typedef __attribute__((ext_vector_type(4)))  unsigned int  v4u;
typedef __attribute__((ext_vector_type(4)))  unsigned int  u32x4;
typedef __attribute__((ext_vector_type(8)))  int           i32x8;
typedef __attribute__((ext_vector_type(4)))  int           i32x4;

union Frag { v4u u[2]; v16bf bf; };   // 32 bytes: 16 bf16 = 8 VGPRs

// hardware bf16 convert (RNE)
__device__ __forceinline__ unsigned short f2bf(float f) {
    return __builtin_bit_cast(unsigned short, (__bf16)f);
}
__device__ __forceinline__ unsigned int pk2bf(float lo, float hi) {
#if defined(__has_builtin) && __has_builtin(__builtin_amdgcn_cvt_pk_bf16_f32)
    auto p = __builtin_amdgcn_cvt_pk_bf16_f32(lo, hi);   // v_cvt_pk_bf16_f32
    return __builtin_bit_cast(unsigned int, p);
#else
    return (unsigned int)f2bf(lo) | ((unsigned int)f2bf(hi) << 16);
#endif
}
__device__ __forceinline__ float bf2f(unsigned int h) {
    return __builtin_bit_cast(float, (h & 0xFFFFu) << 16);
}

// ---------------- Tensor Data Mover availability ----------------
#if defined(__has_builtin)
#if __has_builtin(__builtin_amdgcn_tensor_load_to_lds) && __has_builtin(__builtin_amdgcn_s_wait_tensorcnt)
#define HAVE_TDM 1
#endif
#endif
#ifndef HAVE_TDM
#define HAVE_TDM 0
#endif

#if HAVE_TDM
// DMA a [rows x 32] bf16 tile (row stride = lda elems) from global into LDS,
// padded so the LDS row stride is 40 elements (80B): pad_interval=3 (every
// 16 DWORDs = one 64B row), pad_amount=3 (+4 DWORDs = 16B).
__device__ __forceinline__ void tdm_load_tile_bf16(const unsigned short* gsrc,
                                                   unsigned int lds_off,
                                                   int rows, int lda)
{
    unsigned long long ga = (unsigned long long)gsrc;   // byte address
    u32x4 g0;
    g0[0] = 1u;                                          // count=1, user mode
    g0[1] = lds_off;                                     // lds_addr (bytes)
    g0[2] = (unsigned int)(ga & 0xFFFFFFFFull);          // global_addr[31:0]
    g0[3] = (unsigned int)((ga >> 32) & 0x01FFFFFFull)   // global_addr[56:32]
          | (2u << 30);                                  // type=2 ("image")
    i32x8 g1;
    g1[0] = (int)((1u << 16)      // data_size = 1 (2 bytes)
                | (1u << 20)      // pad_enable
                | (3u << 22)      // pad_interval: 16 DWORDs
                | (3u << 25));    // pad_amount: 4 DWORDs
    // tensor_dim0 = tensor_dim1 = 0x40000000 (huge; tiles never go OOB)
    g1[1] = 0;                                   // dim0[15:0]<<16
    g1[2] = 0x4000;                              // dim0[31:16] | dim1[15:0]<<16
    g1[3] = (int)(0x4000u | (32u << 16));        // dim1[31:16] | tile_dim0=32
    g1[4] = rows;                                // tile_dim1 | tile_dim2=0
    g1[5] = lda;                                 // tensor_dim0_stride[31:0]
    g1[6] = 0;                                   // stride0 hi | stride1 lo
    g1[7] = 0;                                   // stride1 hi
    i32x4 gz; gz[0] = 0; gz[1] = 0; gz[2] = 0; gz[3] = 0;
#if __clang_major__ >= 23
    i32x8 gz8; for (int i = 0; i < 8; ++i) gz8[i] = 0;
    __builtin_amdgcn_tensor_load_to_lds(g0, g1, gz, gz, gz8, 0);
#else
    __builtin_amdgcn_tensor_load_to_lds(g0, g1, gz, gz, 0);
#endif
}
#endif

// =====================================================================
// fp32 -> bf16 bulk convert (8 elems/thread, bandwidth bound)
// =====================================================================
__global__ __launch_bounds__(256)
void cvt_bf16_kernel(const float* __restrict__ in, unsigned short* __restrict__ outp)
{
    size_t i = ((size_t)blockIdx.x * 256 + threadIdx.x) * 8;
    float4 a = *(const float4*)(in + i);
    float4 b = *(const float4*)(in + i + 4);
    uint4 o;
    o.x = pk2bf(a.x, a.y); o.y = pk2bf(a.z, a.w);
    o.z = pk2bf(b.x, b.y); o.w = pk2bf(b.z, b.w);
    *(uint4*)(outp + i) = o;
}

// =====================================================================
// WMMA GEMM:  C[M,N] = op(A[M,K] @ W[K,N] + bias)
//   A: fp32 or bf16 (row-major, lda elems). W: fp32 row-major [K,N].
//   Block tile 128x64, KT=32, 256 threads = 8 waves as 4(M) x 2(N),
//   each wave does 2x2 tiles of v_wmma_f32_16x16x32_bf16.
//   bf16 A tiles are DMA'd by the Tensor Data Mover when available.
// =====================================================================
template<bool ABF16, bool OBF16, bool DOGELU>
__global__ __launch_bounds__(256)
void gemm_wmma(const void* __restrict__ Ap, int lda,
               const float* __restrict__ W, const float* __restrict__ bias,
               void* __restrict__ Cp, int N, int K)
{
    __shared__ alignas(16) unsigned short aS[128 * 40];   // A tile, stride 40 (pad)
    __shared__ alignas(16) unsigned short bS[64 * 40];    // B tile transposed bT[n][k]

    const int tid  = threadIdx.x;
    const int lane = tid & 31, wid = tid >> 5;
    const int wm = wid & 3, wn = wid >> 2;
    const int m0 = blockIdx.y * 128, n0 = blockIdx.x * 64;
    const int rsel = lane & 15, ksel = lane >> 4;

    v8f acc[2][2];
    for (int i = 0; i < 2; ++i)
        for (int j = 0; j < 2; ++j)
            for (int e = 0; e < 8; ++e) acc[i][j][e] = 0.0f;

    const float*          Af = (const float*)Ap;
    const unsigned short* Ab = (const unsigned short*)Ap;

    for (int k0 = 0; k0 < K; k0 += 32) {
        // ---- stage A tile 128x32 -> bf16 LDS (stride 40) ----
        if constexpr (!ABF16) {
            // fp32 A (small M=256 GEMMs only): load, convert, store
            float4 va[4];
            #pragma unroll
            for (int i = 0; i < 4; ++i) {
                int f = tid + 256 * i;
                int r = f >> 3, cs = f & 7;
                va[i] = *(const float4*)(Af + (size_t)(m0 + r) * lda + k0 + cs * 4);
            }
            #pragma unroll
            for (int i = 0; i < 4; ++i) {
                int f = tid + 256 * i;
                int r = f >> 3, cs = f & 7;
                uint2 p;
                p.x = pk2bf(va[i].x, va[i].y);
                p.y = pk2bf(va[i].z, va[i].w);
                *(uint2*)(&aS[r * 40 + cs * 4]) = p;
            }
        } else {
#if HAVE_TDM
            // Tensor Data Mover: one DMA per workgroup, issued by wave 0.
            if (wid == 0) {
                tdm_load_tile_bf16(Ab + (size_t)m0 * lda + k0,
                                   (unsigned int)(unsigned long long)(void*)aS,
                                   128, lda);
                __builtin_amdgcn_s_wait_tensorcnt(0);
            }
            __asm__ volatile("" ::: "memory");
#else
            uint4 va[2];
            #pragma unroll
            for (int i = 0; i < 2; ++i) {
                int f = tid + 256 * i;
                int r = f >> 2, s = f & 3;
                va[i] = *(const uint4*)(Ab + (size_t)(m0 + r) * lda + k0 + s * 8);
            }
            #pragma unroll
            for (int i = 0; i < 2; ++i) {
                int f = tid + 256 * i;
                int r = f >> 2, s = f & 3;
                *(uint4*)(&aS[r * 40 + s * 8]) = va[i];
            }
#endif
        }
        // ---- stage B tile 32x64 (fp32 W) transposed -> bT[n][k] bf16 ----
        {
            float wv[8];
            #pragma unroll
            for (int i = 0; i < 8; ++i) {
                int f = tid + 256 * i;
                int kr = f >> 6, nc = f & 63;
                wv[i] = W[(size_t)(k0 + kr) * N + n0 + nc];
            }
            #pragma unroll
            for (int i = 0; i < 8; ++i) {
                int f = tid + 256 * i;
                int kr = f >> 6, nc = f & 63;
                bS[nc * 40 + kr] = f2bf(wv[i]);
            }
        }
        // prefetch next K tile of W (emits global_prefetch_b8)
        if (k0 + 32 < K) {
            if constexpr (!ABF16)
                __builtin_prefetch(Af + (size_t)(m0 + (tid >> 1)) * lda + k0 + 32, 0, 0);
            __builtin_prefetch(W + (size_t)(k0 + 32 + (tid >> 6)) * N + n0 + (tid & 63), 0, 0);
        }
        __syncthreads();

        // ---- fragments per documented CDNA5 layouts ----
        Frag aF[2], bF[2];
        #pragma unroll
        for (int mi = 0; mi < 2; ++mi) {
            int R  = wm * 32 + mi * 16 + rsel;   // A row (M)
            int kb = ksel * 8;                   // lanes 16-31 start at K=8
            aF[mi].u[0] = *(const v4u*)(&aS[R * 40 + kb]);        // K = kb..kb+7
            aF[mi].u[1] = *(const v4u*)(&aS[R * 40 + 16 + kb]);   // K = 16+kb..16+kb+7
        }
        #pragma unroll
        for (int ni = 0; ni < 2; ++ni) {
            int nn = wn * 32 + ni * 16 + rsel;   // B col (N)
            bF[ni].u[0] = *(const v4u*)(&bS[nn * 40 + ksel * 8]);       // K = ksel*8 + 0..7
            bF[ni].u[1] = *(const v4u*)(&bS[nn * 40 + 16 + ksel * 8]);  // K = 16 + ksel*8 + 0..7
        }
        #pragma unroll
        for (int mi = 0; mi < 2; ++mi)
            #pragma unroll
            for (int ni = 0; ni < 2; ++ni)
                acc[mi][ni] = __builtin_amdgcn_wmma_f32_16x16x32_bf16(
                    false, aF[mi].bf, false, bF[ni].bf,
                    (short)0, acc[mi][ni], false, false);
        __syncthreads();
    }

    // ---- epilogue: C layout row = ksel*8 + j, col = lane&15 ----
    float*          Cf = (float*)Cp;
    unsigned short* Cb = (unsigned short*)Cp;
    #pragma unroll
    for (int mi = 0; mi < 2; ++mi) {
        #pragma unroll
        for (int ni = 0; ni < 2; ++ni) {
            int col  = n0 + wn * 32 + ni * 16 + rsel;
            float bv = bias ? bias[col] : 0.0f;
            #pragma unroll
            for (int j = 0; j < 8; ++j) {
                int row = m0 + wm * 32 + mi * 16 + ksel * 8 + j;
                float v = acc[mi][ni][j] + bv;
                if (DOGELU) v = 0.5f * v * (1.0f + erff(v * 0.70710678118f));
                if (OBF16) Cb[(size_t)row * N + col] = f2bf(v);
                else       Cf[(size_t)row * N + col] = v;
            }
        }
    }
}

// =====================================================================
// broadcast query tokens: q[(b*4+t)*512+d] = qt[t*512+d]
// =====================================================================
__global__ __launch_bounds__(256)
void init_q_kernel(const float* __restrict__ qt, float* __restrict__ q)
{
    int i = blockIdx.x * 256 + threadIdx.x;   // 131072 total
    int t = (i >> 9) & 3;
    q[i] = qt[t * 512 + (i & 511)];
}

// =====================================================================
// cross-attention core: one block per batch, one wave per head.
// qh fp32 [B*4,512]; K,V bf16 [B*512,512]; out fp32 [B*4,512]
// =====================================================================
__global__ __launch_bounds__(256)
void attn_kernel(const float* __restrict__ qh,
                 const unsigned short* __restrict__ Kb,
                 const unsigned short* __restrict__ Vb,
                 float* __restrict__ out)
{
    __shared__ float qs[4 * 512];
    __shared__ float sc[8 * 4 * 512];   // per-head scores/probs (64KB)
    const int b = blockIdx.x;
    const int tid = threadIdx.x, lane = tid & 31, h = tid >> 5;

    for (int i = tid; i < 2048; i += 256) qs[i] = qh[(size_t)b * 2048 + i];
    __syncthreads();

    // scores s[t][j] = q.k / sqrt(64)
    for (int t = 0; t < 4; ++t) {
        const float* qrow = &qs[t * 512 + h * 64];
        for (int jj = 0; jj < 16; ++jj) {
            int j = jj * 32 + lane;
            const uint4* kp = (const uint4*)(Kb + ((size_t)(b * 512 + j)) * 512 + h * 64);
            float a = 0.0f;
            #pragma unroll
            for (int dd = 0; dd < 8; ++dd) {
                uint4 kv = kp[dd];
                const float* q8 = qrow + dd * 8;
                a += bf2f(kv.x) * q8[0] + bf2f(kv.x >> 16) * q8[1]
                   + bf2f(kv.y) * q8[2] + bf2f(kv.y >> 16) * q8[3]
                   + bf2f(kv.z) * q8[4] + bf2f(kv.z >> 16) * q8[5]
                   + bf2f(kv.w) * q8[6] + bf2f(kv.w >> 16) * q8[7];
            }
            sc[(h * 4 + t) * 512 + j] = a * 0.125f;
        }
    }
    // wave-local softmax over S=512 (wave32 shfl reductions)
    for (int t = 0; t < 4; ++t) {
        float* s = &sc[(h * 4 + t) * 512];
        float m = -3.402823466e38f;
        for (int jj = 0; jj < 16; ++jj) m = fmaxf(m, s[jj * 32 + lane]);
        for (int off = 16; off; off >>= 1) m = fmaxf(m, __shfl_xor(m, off, 32));
        float e[16], sum = 0.0f;
        for (int jj = 0; jj < 16; ++jj) { e[jj] = __expf(s[jj * 32 + lane] - m); sum += e[jj]; }
        for (int off = 16; off; off >>= 1) sum += __shfl_xor(sum, off, 32);
        float inv = 1.0f / sum;
        for (int jj = 0; jj < 16; ++jj) s[jj * 32 + lane] = e[jj] * inv;
    }
    // out[t][d] = sum_j p[j] * V[j][d]; lane owns 2 dims
    const int d0 = lane * 2;
    for (int t = 0; t < 4; ++t) {
        const float* p = &sc[(h * 4 + t) * 512];
        float a0 = 0.0f, a1 = 0.0f;
        for (int j = 0; j < 512; ++j) {
            float pj = p[j];
            unsigned pv = *(const unsigned*)(Vb + ((size_t)(b * 512 + j)) * 512 + h * 64 + d0);
            a0 += pj * bf2f(pv);
            a1 += pj * bf2f(pv >> 16);
        }
        out[(size_t)(b * 4 + t) * 512 + h * 64 + d0]     = a0;
        out[(size_t)(b * 4 + t) * 512 + h * 64 + d0 + 1] = a1;
    }
}

// =====================================================================
// out = LayerNorm(x + d) * g + b   (one block per row, HID=512)
// =====================================================================
__global__ __launch_bounds__(256)
void add_ln_kernel(const float* __restrict__ x, const float* __restrict__ d,
                   const float* __restrict__ g, const float* __restrict__ bb,
                   float* __restrict__ out)
{
    __shared__ float red[256];
    const int row = blockIdx.x, tid = threadIdx.x;
    float v0 = x[(size_t)row * 512 + tid]       + d[(size_t)row * 512 + tid];
    float v1 = x[(size_t)row * 512 + 256 + tid] + d[(size_t)row * 512 + 256 + tid];
    red[tid] = v0 + v1; __syncthreads();
    for (int off = 128; off; off >>= 1) { if (tid < off) red[tid] += red[tid + off]; __syncthreads(); }
    float mu = red[0] * (1.0f / 512.0f);
    __syncthreads();
    float e0 = v0 - mu, e1 = v1 - mu;
    red[tid] = e0 * e0 + e1 * e1; __syncthreads();
    for (int off = 128; off; off >>= 1) { if (tid < off) red[tid] += red[tid + off]; __syncthreads(); }
    float inv = rsqrtf(red[0] * (1.0f / 512.0f) + 1e-5f);
    out[(size_t)row * 512 + tid]       = e0 * inv * g[tid]       + bb[tid];
    out[(size_t)row * 512 + 256 + tid] = e1 * inv * g[256 + tid] + bb[256 + tid];
}

// =====================================================================
// residual quantization: one block per (b,t) row; thread c owns code c.
// reconstructed = q - final_residual.
// =====================================================================
__global__ __launch_bounds__(256)
void rq_kernel(const float* __restrict__ q, const float* __restrict__ cbks,
               float* __restrict__ out_ids, float* __restrict__ out_rec)
{
    __shared__ float res[512];
    __shared__ float bv[256];
    __shared__ int   bi[256];
    __shared__ int   ids[3];
    const int row = blockIdx.x, tid = threadIdx.x;
    res[tid]       = q[(size_t)row * 512 + tid];
    res[tid + 256] = q[(size_t)row * 512 + 256 + tid];
    __syncthreads();
    for (int l = 0; l < 3; ++l) {
        const float* c = cbks + ((size_t)l * 256 + tid) * 512;
        float dot = 0.0f, n2 = 0.0f;
        for (int dpos = 0; dpos < 512; dpos += 4) {
            float4 cv = *(const float4*)(c + dpos);
            dot += cv.x * res[dpos] + cv.y * res[dpos + 1] + cv.z * res[dpos + 2] + cv.w * res[dpos + 3];
            n2  += cv.x * cv.x + cv.y * cv.y + cv.z * cv.z + cv.w * cv.w;
        }
        bv[tid] = n2 - 2.0f * dot; bi[tid] = tid;
        __syncthreads();
        for (int off = 128; off; off >>= 1) {
            if (tid < off) {
                float ov = bv[tid + off]; int oi = bi[tid + off];
                if (ov < bv[tid] || (ov == bv[tid] && oi < bi[tid])) { bv[tid] = ov; bi[tid] = oi; }
            }
            __syncthreads();
        }
        int idx = bi[0];
        if (tid == 0) ids[l] = idx;
        const float* cw = cbks + ((size_t)l * 256 + idx) * 512;
        __syncthreads();
        res[tid]       -= cw[tid];
        res[tid + 256] -= cw[tid + 256];
        __syncthreads();
    }
    if (tid < 3) out_ids[row * 3 + tid] = (float)ids[tid];
    out_rec[(size_t)row * 512 + tid]       = q[(size_t)row * 512 + tid]       - res[tid];
    out_rec[(size_t)row * 512 + 256 + tid] = q[(size_t)row * 512 + 256 + tid] - res[tid + 256];
}

// =====================================================================
extern "C" void kernel_launch(void* const* d_in, const int* in_sizes, int n_in,
                              void* d_out, int out_size, void* d_ws, size_t ws_size,
                              hipStream_t stream)
{
    (void)in_sizes; (void)n_in; (void)out_size; (void)ws_size;
    const float* mm  = (const float*)d_in[0];
    const float* pW  = (const float*)d_in[1];
    const float* pb  = (const float*)d_in[2];
    const float* qt  = (const float*)d_in[3];
    const float* Wq  = (const float*)d_in[4];
    const float* bq  = (const float*)d_in[5];
    const float* Wk  = (const float*)d_in[6];
    const float* bk  = (const float*)d_in[7];
    const float* Wv  = (const float*)d_in[8];
    const float* bvv = (const float*)d_in[9];
    const float* Wo  = (const float*)d_in[10];
    const float* bo  = (const float*)d_in[11];
    const float* W1  = (const float*)d_in[12];
    const float* b1  = (const float*)d_in[13];
    const float* W2  = (const float*)d_in[14];
    const float* b2  = (const float*)d_in[15];
    const float* lng = (const float*)d_in[16];
    const float* lnb = (const float*)d_in[17];
    const float* cbk = (const float*)d_in[18];
    float* out = (float*)d_out;

    // ---- workspace layout (~171 MB) ----
    char* ws = (char*)d_ws;
    unsigned short* mmbf = (unsigned short*)ws;                      // 32768x1024 bf16 = 64MB
    unsigned short* xbf  = (unsigned short*)(ws +  67108864);        // 32768x512  bf16 = 32MB
    unsigned short* Kbf  = (unsigned short*)(ws + 100663296);        // 32MB
    unsigned short* Vbf  = (unsigned short*)(ws + 134217728);        // 32MB
    float* qbuf = (float*)(ws + 167772160);                          // 256x512 f32
    float* qh   = qbuf + 131072;
    float* attn = qh   + 131072;
    float* tmp  = attn + 131072;
    unsigned short* h1 = (unsigned short*)(tmp + 131072);            // 256x2048 bf16

    init_q_kernel<<<512, 256, 0, stream>>>(qt, qbuf);

    // pre-convert multimodal features to bf16 (33.5M elems, 8/thread)
    cvt_bf16_kernel<<<16384, 256, 0, stream>>>(mm, mmbf);

    // x = bf16(MM @ projW + b) : M=32768 K=1024 N=512 (TDM-fed A tiles)
    gemm_wmma<true, true, false><<<dim3(8, 256), 256, 0, stream>>>(mmbf, 1024, pW, pb, xbf, 512, 1024);

    for (int i = 0; i < 4; ++i) {
        const float* Wqi = Wq + (size_t)i * 262144; const float* bqi = bq + (size_t)i * 512;
        const float* Wki = Wk + (size_t)i * 262144; const float* bki = bk + (size_t)i * 512;
        const float* Wvi = Wv + (size_t)i * 262144; const float* bvi = bvv + (size_t)i * 512;
        const float* Woi = Wo + (size_t)i * 262144; const float* boi = bo + (size_t)i * 512;
        const float* W1i = W1 + (size_t)i * 1048576; const float* b1i = b1 + (size_t)i * 2048;
        const float* W2i = W2 + (size_t)i * 1048576; const float* b2i = b2 + (size_t)i * 512;
        const float* gi  = lng + (size_t)i * 512;    const float* bi_ = lnb + (size_t)i * 512;

        // K,V = bf16(x @ W + b) : M=32768 K=512 N=512  (TDM-fed A tiles)
        gemm_wmma<true,  true,  false><<<dim3(8, 256), 256, 0, stream>>>(xbf, 512, Wki, bki, Kbf, 512, 512);
        gemm_wmma<true,  true,  false><<<dim3(8, 256), 256, 0, stream>>>(xbf, 512, Wvi, bvi, Vbf, 512, 512);
        // qh = q @ Wq + bq : M=256
        gemm_wmma<false, false, false><<<dim3(8, 2),   256, 0, stream>>>(qbuf, 512, Wqi, bqi, qh, 512, 512);
        // attention
        attn_kernel<<<64, 256, 0, stream>>>(qh, Kbf, Vbf, attn);
        // out proj
        gemm_wmma<false, false, false><<<dim3(8, 2),   256, 0, stream>>>(attn, 512, Woi, boi, tmp, 512, 512);
        add_ln_kernel<<<256, 256, 0, stream>>>(qbuf, tmp, gi, bi_, qbuf);
        // FFN
        gemm_wmma<false, true,  true ><<<dim3(32, 2),  256, 0, stream>>>(qbuf, 512, W1i, b1i, h1, 2048, 512);
        gemm_wmma<true,  false, false><<<dim3(8, 2),   256, 0, stream>>>(h1, 2048, W2i, b2i, tmp, 512, 2048);
        add_ln_kernel<<<256, 256, 0, stream>>>(qbuf, tmp, gi, bi_, qbuf);
    }

    // residual quantization -> d_out = [ids(768) | reconstructed(131072)]
    rq_kernel<<<256, 256, 0, stream>>>(qbuf, cbk, out, out + 768);
}
